// GNNEncoder_86071144611862
// MI455X (gfx1250) — matrix operand order repaired and checked by reading, not compiled, not run on hardware
//
#include <hip/hip_runtime.h>
#include <hip/hip_bf16.h>

// ---------------------------------------------------------------------------
// GraphConv x3 (PyG GraphConv, aggr='add') for MI455X / gfx1250.
//   agg = segment_sum(h[src], dst);  out = agg@w_rel + h@w_root + b; relu(1,2)
// GEMMs run on v_wmma_f32_16x16x32_bf16 with hi/lo bf16 split (fp32-accurate).
// Scatter uses global_atomic_add_f32 (unsafeAtomicAdd), float4 per lane.
// ---------------------------------------------------------------------------

typedef __attribute__((ext_vector_type(16))) __bf16 v16bf;
typedef __attribute__((ext_vector_type(8)))  float  v8f;

#define D_FEAT 128

__device__ __forceinline__ v8f wmma_bf16(v16bf a, v16bf b, v8f c) {
  return __builtin_amdgcn_wmma_f32_16x16x32_bf16(
      /*neg_a=*/false, a, /*neg_b=*/false, b,
      /*c_mod=*/(short)0, c, /*reuse_a=*/false, /*reuse_b=*/false);
}

__device__ __forceinline__ void cvt_hilo(float f, __bf16& hi, __bf16& lo) {
  hi = (__bf16)f;
  lo = (__bf16)(f - (float)hi);
}

// ---------------------------------------------------------------------------
// Weight prep: split 128x128 fp32 (row-major K x N) into hi/lo bf16 fragments
// laid out so each lane's v16bf load is one contiguous 32B read.
// Layout: [ntile(8)][kchunk(4)][hl(2)][lane(32)][j(16)]  (32768 bf16 = 64KB)
// B-fragment mapping (16x16x32): n = ntile*16 + (lane&15),
//                                k = kchunk*32 + (j<8 ? 8*(lane>>4) + j : 16 + 8*(lane>>4) + (j-8))
// ---------------------------------------------------------------------------
__global__ __launch_bounds__(256)
void prep_w(const float* __restrict__ w, unsigned short* __restrict__ wf) {
  int t = blockIdx.x * blockDim.x + threadIdx.x;   // 0..16383
  if (t >= 16384) return;
  int j     = t & 15;
  int lane  = (t >> 4) & 31;
  int kc    = (t >> 9) & 3;
  int ntile = t >> 11;
  int k = kc * 32 + (j < 8 ? 8 * (lane >> 4) + j : 16 + 8 * (lane >> 4) + (j - 8));
  int n = ntile * 16 + (lane & 15);
  float v = w[k * D_FEAT + n];
  __bf16 hi, lo;
  cvt_hilo(v, hi, lo);
  size_t base = (((size_t)(ntile * 4 + kc) * 2) * 32 + lane) * 16 + j;
  wf[base]       = __builtin_bit_cast(unsigned short, hi);
  wf[base + 512] = __builtin_bit_cast(unsigned short, lo);  // hl=1 slot
}

// ---------------------------------------------------------------------------
// Zero fill (graph-capture-safe replacement for hipMemsetAsync)
// ---------------------------------------------------------------------------
__global__ __launch_bounds__(256)
void zero_buf(float4* __restrict__ p, int n4) {
  int i = blockIdx.x * blockDim.x + threadIdx.x;
  if (i < n4) p[i] = make_float4(0.f, 0.f, 0.f, 0.f);
}

// ---------------------------------------------------------------------------
// Edge scatter: one wave per edge, lane handles 4 contiguous features.
// agg[dst] += h[src]  via non-returning f32 global atomics.
// ---------------------------------------------------------------------------
__global__ __launch_bounds__(256)
void scatter_edges(const float* __restrict__ h, const int* __restrict__ src,
                   const int* __restrict__ dst, float* __restrict__ agg,
                   int n_edges) {
  long long gid = (long long)blockIdx.x * blockDim.x + threadIdx.x;
  int edge = (int)(gid >> 5);
  int lane = (int)(gid & 31);
  if (edge >= n_edges) return;
  int s = src[edge];
  int d = dst[edge];
  const float4 v = *(const float4*)(h + (size_t)s * D_FEAT + lane * 4);
  float* p = agg + (size_t)d * D_FEAT + lane * 4;
  unsafeAtomicAdd(p + 0, v.x);
  unsafeAtomicAdd(p + 1, v.y);
  unsafeAtomicAdd(p + 2, v.z);
  unsafeAtomicAdd(p + 3, v.w);
}

// ---------------------------------------------------------------------------
// Fused dual GEMM: out = relu?( agg @ w_rel + h @ w_root + b )
// Block = 256 threads (8 waves) per 16-node row block; wave = one 16-col tile.
// Per wave: 2 matrices x 4 K-chunks x 3 (hi*hi, hi*lo, lo*hi) = 24 WMMAs.
// A fragment (16x32 bf16): m = lane&15, g = lane>>4,
//   j in [0,8)  -> K = c*32 + 8g + j
//   j in [8,16) -> K = c*32 + 16 + 8g + (j-8)
// C/D: col = ntile*16 + (lane&15); VGPR r -> row = node_base + r + 8g.
// ---------------------------------------------------------------------------
__global__ __launch_bounds__(256)
void gemm_dual(const float* __restrict__ Aagg, const float* __restrict__ Ah,
               const unsigned short* __restrict__ wfRel,
               const unsigned short* __restrict__ wfRoot,
               const float* __restrict__ bias, float* __restrict__ out,
               int do_relu) {
  const int lane      = threadIdx.x & 31;
  const int ntile     = threadIdx.x >> 5;     // 0..7 column tile
  const int node_base = blockIdx.x * 16;
  const int m = lane & 15;
  const int g = lane >> 4;
  const int ncol = ntile * 16 + m;

  v8f acc;
  const float bv = bias[ncol];
#pragma unroll
  for (int r = 0; r < 8; ++r) acc[r] = bv;

#pragma unroll
  for (int mat = 0; mat < 2; ++mat) {
    const float* A = mat ? Ah : Aagg;
    const unsigned short* WF = mat ? wfRoot : wfRel;
    const float* rowA = A + (size_t)(node_base + m) * D_FEAT;
#pragma unroll
    for (int c = 0; c < 4; ++c) {
      v16bf a_hi, a_lo;
#pragma unroll
      for (int j = 0; j < 8; ++j) {
        float f = rowA[c * 32 + 8 * g + j];
        __bf16 hi, lo;
        cvt_hilo(f, hi, lo);
        a_hi[j] = hi; a_lo[j] = lo;
      }
#pragma unroll
      for (int j = 0; j < 8; ++j) {
        float f = rowA[c * 32 + 16 + 8 * g + j];
        __bf16 hi, lo;
        cvt_hilo(f, hi, lo);
        a_hi[8 + j] = hi; a_lo[8 + j] = lo;
      }
      const unsigned short* bp =
          WF + (((size_t)(ntile * 4 + c) * 2 + 0) * 32 + lane) * 16;
      v16bf b_hi = *(const v16bf*)bp;
      v16bf b_lo = *(const v16bf*)(bp + 512);   // hl=1 slot
      acc = wmma_bf16(a_hi, b_hi, acc);
      acc = wmma_bf16(a_hi, b_lo, acc);
      acc = wmma_bf16(a_lo, b_hi, acc);
    }
  }

  if (do_relu) {
#pragma unroll
    for (int r = 0; r < 8; ++r) acc[r] = fmaxf(acc[r], 0.0f);
  }
#pragma unroll
  for (int r = 0; r < 8; ++r) {
    int row = node_base + r + 8 * g;
    out[(size_t)row * D_FEAT + ncol] = acc[r];
  }
}

// ---------------------------------------------------------------------------
// Launch
// ---------------------------------------------------------------------------
extern "C" void kernel_launch(void* const* d_in, const int* in_sizes, int n_in,
                              void* d_out, int out_size, void* d_ws, size_t ws_size,
                              hipStream_t stream) {
  const float* x       = (const float*)d_in[0];
  const int*   ei      = (const int*)d_in[1];     // int32 (JAX default x64-off)
  const float* w_rel1  = (const float*)d_in[2];
  const float* w_root1 = (const float*)d_in[3];
  const float* b1      = (const float*)d_in[4];
  const float* w_rel2  = (const float*)d_in[5];
  const float* w_root2 = (const float*)d_in[6];
  const float* b2      = (const float*)d_in[7];
  const float* w_rel3  = (const float*)d_in[8];
  const float* w_root3 = (const float*)d_in[9];
  const float* b3      = (const float*)d_in[10];

  const int n_nodes = in_sizes[0] / D_FEAT;   // 100000
  const int n_edges = in_sizes[1] / 2;        // 1600000
  const int* src = ei;
  const int* dst = ei + n_edges;

  // Workspace layout: [6 x 64KB weight fragments][agg][h2]
  char* ws = (char*)d_ws;
  unsigned short* wf = (unsigned short*)ws;
  const size_t WFRAG_ELEMS = 32768;                       // per matrix
  const size_t wf_bytes = 6 * WFRAG_ELEMS * sizeof(unsigned short);
  const size_t node_bytes = (size_t)n_nodes * D_FEAT * sizeof(float);
  float* agg = (float*)(ws + wf_bytes);
  float* h2  = (float*)(ws + wf_bytes + node_bytes);
  float* h1  = (float*)d_out;                             // reused, fully rewritten
  float* outp = (float*)d_out;

  const float* wmats[6] = {w_rel1, w_root1, w_rel2, w_root2, w_rel3, w_root3};
  for (int i = 0; i < 6; ++i)
    prep_w<<<64, 256, 0, stream>>>(wmats[i], wf + (size_t)i * WFRAG_ELEMS);

  const int nf4      = n_nodes * D_FEAT / 4;
  const int zblocks  = (nf4 + 255) / 256;
  const long long st = (long long)n_edges * 32;
  const int sblocks  = (int)((st + 255) / 256);
  const int gblocks  = (n_nodes + 15) / 16;               // 6250

  // ---- layer 1: h1 = relu(agg(x) @ w_rel1 + x @ w_root1 + b1)
  zero_buf<<<zblocks, 256, 0, stream>>>((float4*)agg, nf4);
  scatter_edges<<<sblocks, 256, 0, stream>>>(x, src, dst, agg, n_edges);
  gemm_dual<<<gblocks, 256, 0, stream>>>(agg, x, wf + 0 * WFRAG_ELEMS,
                                         wf + 1 * WFRAG_ELEMS, b1, h1, 1);
  // ---- layer 2: h2 = relu(agg(h1) @ w_rel2 + h1 @ w_root2 + b2)
  zero_buf<<<zblocks, 256, 0, stream>>>((float4*)agg, nf4);
  scatter_edges<<<sblocks, 256, 0, stream>>>(h1, src, dst, agg, n_edges);
  gemm_dual<<<gblocks, 256, 0, stream>>>(agg, h1, wf + 2 * WFRAG_ELEMS,
                                         wf + 3 * WFRAG_ELEMS, b2, h2, 1);
  // ---- layer 3: out = agg(h2) @ w_rel3 + h2 @ w_root3 + b3
  zero_buf<<<zblocks, 256, 0, stream>>>((float4*)agg, nf4);
  scatter_edges<<<sblocks, 256, 0, stream>>>(h2, src, dst, agg, n_edges);
  gemm_dual<<<gblocks, 256, 0, stream>>>(agg, h2, wf + 4 * WFRAG_ELEMS,
                                         wf + 5 * WFRAG_ELEMS, b3, outp, 0);
}
